// SelfAttentionGraphPooling_13056700579977
// MI455X (gfx1250) — matrix operand order repaired and checked by reading, not compiled, not run on hardware
//
#include <hip/hip_runtime.h>
#include <hip/hip_bf16.h>
#include <math.h>

typedef __attribute__((ext_vector_type(16))) _Float16 v16h;
typedef __attribute__((ext_vector_type(8)))  float    v8f;

#define B_    8
#define N_    50000
#define F_    256
#define HID_  256
#define K_    25000
#define NBINS 65536
#define LDS_STRIDE 272   // 256 + 16 f16 pad -> 544B rows, 32B aligned, conflict-free-ish

// Hardware tanh (gfx1250 TRANS op). Builtin when available, else inline asm.
// ISA: TRANS result needs 1 independent op before use -> embed v_nop in asm path.
__device__ __forceinline__ float fast_tanh(float x) {
#if __has_builtin(__builtin_amdgcn_tanhf)
    return __builtin_amdgcn_tanhf(x);
#else
    float r;
    asm volatile("v_tanh_f32 %0, %1\n\tv_nop" : "=v"(r) : "v"(x));
    return r;
#endif
}

// ---------------------------------------------------------------------------
// Kernel 1: scores = tanh(h @ w1 + b1) @ w2 + b2, written as sortable u32 keys
// (ascending key == descending score). One wave handles 16 nodes; WMMA
// f32_16x16x32_f16 over K=256 (8 steps) x 16 hidden column tiles.
// ---------------------------------------------------------------------------
__global__ __launch_bounds__(256) void k_scores(
    const float* __restrict__ h, const float* __restrict__ w1,
    const float* __restrict__ b1, const float* __restrict__ w2,
    const float* __restrict__ b2, unsigned int* __restrict__ keys)
{
    extern __shared__ _Float16 lds[];   // 16 cols x LDS_STRIDE f16 (w1 tile, transposed)
    const int b    = blockIdx.y;
    const int t    = threadIdx.x;
    const int lane = t & 31;
    const int wave = t >> 5;
    const int rowBase = blockIdx.x * 128 + wave * 16;

    const int myRow  = rowBase + (lane & 15);
    const int rowClp = (myRow < N_) ? myRow : (N_ - 1);
    const float* hrow = h + ((size_t)b * N_ + rowClp) * F_;

    // ---- A fragments (16x32 f16), contiguous k = 16*(lane/16) + s lane layout, cached in registers ----
    const int koffA = (lane & 16) ? 16 : 0;
    v16h a[8];
#pragma unroll
    for (int ks = 0; ks < 8; ++ks) {
        const float* p0 = hrow + ks * 32 + koffA;
        float4 q0 = *(const float4*)(p0);
        float4 q1 = *(const float4*)(p0 + 4);
        float4 q2 = *(const float4*)(p0 + 8);
        float4 q3 = *(const float4*)(p0 + 12);
        v16h af;
        af[0]=(_Float16)q0.x; af[1]=(_Float16)q0.y; af[2]=(_Float16)q0.z; af[3]=(_Float16)q0.w;
        af[4]=(_Float16)q1.x; af[5]=(_Float16)q1.y; af[6]=(_Float16)q1.z; af[7]=(_Float16)q1.w;
        af[8]=(_Float16)q2.x; af[9]=(_Float16)q2.y; af[10]=(_Float16)q2.z; af[11]=(_Float16)q2.w;
        af[12]=(_Float16)q3.x; af[13]=(_Float16)q3.y; af[14]=(_Float16)q3.z; af[15]=(_Float16)q3.w;
        a[ks] = af;
    }

    const int col   = lane & 15;
    const int koffB = (lane & 16) ? 16 : 0;
    float sacc[8];
#pragma unroll
    for (int j = 0; j < 8; ++j) sacc[j] = 0.0f;

    for (int nt = 0; nt < 16; ++nt) {
        // ---- stage w1 column tile (16 cols x 256 k), transposed, into LDS ----
        __syncthreads();
        {
            const int cc   = t >> 4;          // local col 0..15
            const int k0   = (t & 15) * 16;   // 16 k values
            const int gcol = nt * 16 + cc;
#pragma unroll
            for (int kk = 0; kk < 16; ++kk)
                lds[cc * LDS_STRIDE + k0 + kk] =
                    (_Float16)w1[(size_t)(k0 + kk) * HID_ + gcol];
        }
        __syncthreads();

        const float bias = b1[nt * 16 + col];
        v8f c;
#pragma unroll
        for (int j = 0; j < 8; ++j) c[j] = bias;

#pragma unroll
        for (int ks = 0; ks < 8; ++ks) {
            const v16h bf = *(const v16h*)(lds + col * LDS_STRIDE + ks * 32 + koffB);
            c = __builtin_amdgcn_wmma_f32_16x16x32_f16(
                    false, a[ks], false, bf, (short)0, c, false, false);
        }

        const float w2v = w2[nt * 16 + col];
#pragma unroll
        for (int j = 0; j < 8; ++j) sacc[j] += w2v * fast_tanh(c[j]);
    }

    // ---- reduce across the 16 lanes sharing each row half ----
#pragma unroll
    for (int j = 0; j < 8; ++j) {
        float s = sacc[j];
        s += __shfl_xor(s, 1, 32);
        s += __shfl_xor(s, 2, 32);
        s += __shfl_xor(s, 4, 32);
        s += __shfl_xor(s, 8, 32);
        sacc[j] = s;
    }
    if ((lane & 15) == 0) {
        const int mbase = (lane & 16) ? 8 : 0;
        const float bb = b2[0];
#pragma unroll
        for (int j = 0; j < 8; ++j) {
            const int node = rowBase + mbase + j;
            if (node < N_) {
                unsigned int u = __float_as_uint(sacc[j] + bb);
                u ^= (u & 0x80000000u) ? 0xFFFFFFFFu : 0x80000000u; // ascending = bigger score
                keys[(size_t)b * N_ + node] = ~u;                   // ascending = smaller score rank
            }
        }
    }
}

// ---------------------------------------------------------------------------
// Selection pipeline (exact top-K with jax tie semantics: ties -> lower index)
// ---------------------------------------------------------------------------
__global__ void k_clear(unsigned int* __restrict__ p, size_t n) {
    size_t i = (size_t)blockIdx.x * blockDim.x + threadIdx.x;
    if (i < n) p[i] = 0u;
}

__global__ void k_hist(const unsigned int* __restrict__ keys,
                       unsigned int* __restrict__ hist) {
    const int gid = blockIdx.x * blockDim.x + threadIdx.x;
    const int b   = blockIdx.y;
    if (gid >= N_) return;
    const unsigned int key = keys[(size_t)b * N_ + gid];
    atomicAdd(&hist[(size_t)b * NBINS + (key >> 16)], 1u);
}

__global__ __launch_bounds__(256) void k_scan(
    const unsigned int* __restrict__ hist, unsigned int* __restrict__ base) {
    __shared__ unsigned int sh[256];
    __shared__ unsigned int total;
    const int b = blockIdx.x;
    const int t = threadIdx.x;
    if (t == 0) total = 0u;
    __syncthreads();
    for (int chunk = 0; chunk < NBINS / 256; ++chunk) {
        const int i = chunk * 256 + t;
        const unsigned int v = hist[(size_t)b * NBINS + i];
        sh[t] = v;
        __syncthreads();
        for (int off = 1; off < 256; off <<= 1) {
            const unsigned int add = (t >= off) ? sh[t - off] : 0u;
            __syncthreads();
            sh[t] += add;
            __syncthreads();
        }
        base[(size_t)b * NBINS + i] = total + sh[t] - v; // exclusive
        __syncthreads();
        if (t == 0) total += sh[255];
        __syncthreads();
    }
}

__global__ void k_scatter(const unsigned int* __restrict__ keys,
                          const unsigned int* __restrict__ base,
                          unsigned int* __restrict__ cnt,
                          unsigned int* __restrict__ bucketKeys,
                          unsigned int* __restrict__ bucketIdx) {
    const int gid = blockIdx.x * blockDim.x + threadIdx.x;
    const int b   = blockIdx.y;
    if (gid >= N_) return;
    const unsigned int key = keys[(size_t)b * N_ + gid];
    const unsigned int bin = key >> 16;
    const unsigned int pos = base[(size_t)b * NBINS + bin] +
                             atomicAdd(&cnt[(size_t)b * NBINS + bin], 1u);
    bucketKeys[(size_t)b * N_ + pos] = key;
    bucketIdx [(size_t)b * N_ + pos] = (unsigned int)gid;
}

__global__ void k_rank(const unsigned int* __restrict__ bucketKeys,
                       const unsigned int* __restrict__ bucketIdx,
                       const unsigned int* __restrict__ base,
                       const unsigned int* __restrict__ cnt,
                       unsigned int* __restrict__ outIdx) {
    const int gid = blockIdx.x * blockDim.x + threadIdx.x;
    const int b   = blockIdx.y;
    if (gid >= N_) return;
    const size_t o = (size_t)b * N_ + gid;
    const unsigned int key = bucketKeys[o];
    const unsigned int idx = bucketIdx[o];
    const unsigned int bin = key >> 16;
    const unsigned int start = base[(size_t)b * NBINS + bin];
    const unsigned int c     = cnt [(size_t)b * NBINS + bin];
    const unsigned int* kp = bucketKeys + (size_t)b * N_ + start;
    const unsigned int* ip = bucketIdx  + (size_t)b * N_ + start;
    unsigned int r = start;
    for (unsigned int j = 0; j < c; ++j) {
        const unsigned int kj = kp[j];
        const unsigned int ij = ip[j];
        r += (kj < key) || (kj == key && ij < idx);
    }
    if (r < K_) outIdx[(size_t)b * K_ + r] = idx;
}

__global__ __launch_bounds__(256) void k_gather(const float* __restrict__ h,
                                                const unsigned int* __restrict__ outIdx,
                                                float* __restrict__ out) {
    const int b  = blockIdx.y;
    const int r  = blockIdx.x * 4 + (threadIdx.x >> 6);
    const int c4 = threadIdx.x & 63;
    if (r >= K_) return;
    const unsigned int idx = outIdx[(size_t)b * K_ + r];
    const float4* src = (const float4*)(h   + ((size_t)b * N_ + idx) * F_);
    float4*       dst = (float4*)      (out + ((size_t)b * K_ + r)   * F_);
    dst[c4] = src[c4];
}

// ---------------------------------------------------------------------------
extern "C" void kernel_launch(void* const* d_in, const int* in_sizes, int n_in,
                              void* d_out, int out_size, void* d_ws, size_t ws_size,
                              hipStream_t stream) {
    const float* h  = (const float*)d_in[0];
    const float* w1 = (const float*)d_in[1];
    const float* b1 = (const float*)d_in[2];
    const float* w2 = (const float*)d_in[3];
    const float* b2 = (const float*)d_in[4];
    float* out = (float*)d_out;

    char* ws = (char*)d_ws;
    unsigned int* keys       = (unsigned int*)ws;  ws += (size_t)B_ * N_ * 4;
    unsigned int* bucketKeys = (unsigned int*)ws;  ws += (size_t)B_ * N_ * 4;
    unsigned int* bucketIdx  = (unsigned int*)ws;  ws += (size_t)B_ * N_ * 4;
    unsigned int* hist       = (unsigned int*)ws;  ws += (size_t)B_ * NBINS * 4;
    unsigned int* cnt        = (unsigned int*)ws;  ws += (size_t)B_ * NBINS * 4;  // contiguous after hist
    unsigned int* base       = (unsigned int*)ws;  ws += (size_t)B_ * NBINS * 4;
    unsigned int* outIdx     = (unsigned int*)ws;  ws += (size_t)B_ * K_ * 4;

    // 1) scores -> sortable keys (WMMA)
    dim3 gScores((N_ + 127) / 128, B_);
    k_scores<<<gScores, 256, 16 * LDS_STRIDE * sizeof(_Float16), stream>>>(
        h, w1, b1, w2, b2, keys);

    // 2) clear hist + cnt (contiguous)
    const size_t clearN = (size_t)B_ * NBINS * 2;
    k_clear<<<(unsigned)((clearN + 255) / 256), 256, 0, stream>>>(hist, clearN);

    // 3) histogram on top 16 key bits
    dim3 gN((N_ + 255) / 256, B_);
    k_hist<<<gN, 256, 0, stream>>>(keys, hist);

    // 4) per-batch exclusive scan over 65536 bins
    k_scan<<<B_, 256, 0, stream>>>(hist, base);

    // 5) scatter into bucket order
    k_scatter<<<gN, 256, 0, stream>>>(keys, base, cnt, bucketKeys, bucketIdx);

    // 6) exact rank within bucket -> topk index table
    k_rank<<<gN, 256, 0, stream>>>(bucketKeys, bucketIdx, base, cnt, outIdx);

    // 7) gather rows: out[b, rank, :] = h[b, idx, :]
    dim3 gG((K_ + 3) / 4, B_);
    k_gather<<<gG, 256, 0, stream>>>(h, outIdx, out);
}
